// DHGCN_7851200217522
// MI455X (gfx1250) — matrix operand order repaired and checked by reading, not compiled, not run on hardware
//
#include <hip/hip_runtime.h>
#include <hip/hip_bf16.h>
#include <stddef.h>
#include <stdint.h>

typedef _Float16 h16v __attribute__((ext_vector_type(16)));
typedef _Float16 h8v  __attribute__((ext_vector_type(8)));
typedef float    f8v  __attribute__((ext_vector_type(8)));
typedef float    f4v  __attribute__((ext_vector_type(4)));

#define BM 128
#define BN 128
#define BK 32
#define LDS_LDA 40   // halves per A row in LDS (32 + 8 pad)
#define LDS_LDB 40   // halves per B row in LDS
#define A_TILE (BM * LDS_LDA)
#define B_TILE (BN * LDS_LDB)

// ---------------------------------------------------------------------------
// CDNA5 async global->LDS copy (ASYNCcnt-tracked, bypasses VGPRs).
// GV mode: 64-bit vaddr, saddr=off. LDS destination offset = low 32 bits of
// the generic LDS pointer (flat-aperture mapping).
// ---------------------------------------------------------------------------
__device__ __forceinline__ void async_copy_b128(void* lds_dst, const void* gsrc)
{
    uint32_t loff = (uint32_t)(uintptr_t)lds_dst;
    asm volatile("global_load_async_to_lds_b128 %0, %1, off"
                 :: "v"(loff), "v"(gsrc)
                 : "memory");
}

// ---------------------------------------------------------------------------
// Weight prep: fp32 W[K][N] -> f16 Wt[Npad][Kpad] (transposed, zero padded).
// ---------------------------------------------------------------------------
__global__ __launch_bounds__(256) void convert_transpose_w(
    const float* __restrict__ W, _Float16* __restrict__ Wt,
    int K, int N, int Kpad, int Npad)
{
    int idx = blockIdx.x * blockDim.x + threadIdx.x;
    int total = Kpad * Npad;
    if (idx >= total) return;
    int n = idx / Kpad;
    int k = idx - n * Kpad;
    float v = (k < K && n < N) ? W[(size_t)k * N + n] : 0.0f;
    Wt[idx] = (_Float16)v;
}

// ---------------------------------------------------------------------------
// Fused GEMM + bias + ReLU:  C = relu(A[M,K] @ B[K,N] + bias)
//   A: row-major fp32 (layer 0, converted on the fly) or f16
//   Bt: f16 transposed weights [N][Kpad] (ldb = Kpad), zero padded
//   OUT_F32=false: f16 [M][ldc] out; OUT_F32=true: f32 [M][ldc], n < Nvalid
// Block: 128x128 tile, 256 threads = 8 wave32 waves, wave = 32x64 sub-tile.
// Double-buffered LDS; tile ks+1 streams in via ASYNCcnt-tracked copies while
// tile ks is consumed by WMMAs. Last iteration peeled so the steady-state
// loop is branch-free.
// ---------------------------------------------------------------------------
template <typename AT, bool OUT_F32>
__global__ __launch_bounds__(256) void gemm_bias_relu(
    const AT* __restrict__ A, int lda,
    const _Float16* __restrict__ Bt, int ldb,
    const float* __restrict__ bias,
    void* __restrict__ Cout, int ldc,
    int M, int K, int Nvalid)
{
    __shared__ _Float16 As[2 * A_TILE];
    __shared__ _Float16 Bs[2 * B_TILE];

    const int tid   = threadIdx.x;
    const int lane  = tid & 31;
    const int wave  = tid >> 5;     // 0..7
    const int waveM = wave & 3;     // 4 waves along M (32 rows each)
    const int waveN = wave >> 2;    // 2 waves along N (64 cols each)
    const int blockM0 = blockIdx.x * BM;
    const int blockN0 = blockIdx.y * BN;

    f8v acc[2][4];
#pragma unroll
    for (int i = 0; i < 2; i++)
#pragma unroll
        for (int j = 0; j < 4; j++)
#pragma unroll
            for (int r = 0; r < 8; r++) acc[i][j][r] = 0.0f;

    // Staging: 2 threads per row, 16 contiguous K halves each.
    const int srow  = tid >> 1;           // 0..127
    const int skseg = (tid & 1) * 16;     // 0 or 16

    const int mrow  = blockM0 + srow;
    const int mload = mrow < M ? mrow : (M - 1);          // clamp, no predication
    const AT*       aRow = A  + (size_t)mload * lda + skseg;
    const _Float16* bRow = Bt + (size_t)(blockN0 + srow) * ldb + skseg;
    _Float16* aDst = &As[srow * LDS_LDA + skseg];
    _Float16* bDst = &Bs[srow * LDS_LDB + skseg];

    // Fragment base pointers (this code's contiguous k order, shared by A and B):
    //  A 16x32: lane = row, halves = 16 contiguous K per lane half-group
    //  B 32x16: lane = column, halves = 16 contiguous K per lane half-group
    const int lrow = lane & 15;
    const int lhi  = lane >> 4;
    const _Float16* aFragP = &As[(waveM * 32 + lrow) * LDS_LDA + lhi * 16];
    const _Float16* bFragP = &Bs[(waveN * 64 + lrow) * LDS_LDB + lhi * 16];

    // ---- stage tile ktile into buffer buf ----
    auto stage = [&](int ktile, int buf) {
        const int k0 = ktile * BK;
        _Float16* aD = aDst + buf * A_TILE;
        _Float16* bD = bDst + buf * B_TILE;
        // B: weights, zero-padded to Kpad -> always async, always in range
        async_copy_b128(bD,     bRow + k0);
        async_copy_b128(bD + 8, bRow + k0 + 8);
        if constexpr (sizeof(AT) == 4) {
            // fp32 -> f16 conversion must pass through VGPRs
            if (k0 + BK <= K) {                          // wave-uniform fast path
                const f4v* s4 = (const f4v*)(aRow + k0);
                f4v t0 = s4[0], t1 = s4[1], t2 = s4[2], t3 = s4[3];
                h8v h0, h1;
#pragma unroll
                for (int e = 0; e < 4; e++) {
                    h0[e]     = (_Float16)t0[e];
                    h0[4 + e] = (_Float16)t1[e];
                    h1[e]     = (_Float16)t2[e];
                    h1[4 + e] = (_Float16)t3[e];
                }
                *(h8v*)aD       = h0;
                *(h8v*)(aD + 8) = h1;
            } else {                                     // k-tail: clamp + select
                h8v h0, h1;
#pragma unroll
                for (int i = 0; i < 8; i++) {
                    int ka = k0 + skseg + i;
                    int kb = ka + 8;
                    float va = (float)aRow[(ka < K ? ka : K - 1) - skseg];
                    float vb = (float)aRow[(kb < K ? kb : K - 1) - skseg];
                    h0[i] = (_Float16)(ka < K ? va : 0.0f);
                    h1[i] = (_Float16)(kb < K ? vb : 0.0f);
                }
                *(h8v*)aD       = h0;
                *(h8v*)(aD + 8) = h1;
            }
        } else {
            // f16 activations: K is a multiple of BK by construction
            async_copy_b128(aD,     aRow + k0);
            async_copy_b128(aD + 8, aRow + k0 + 8);
        }
    };

    // ---- consume tile in buffer buf ----
    auto compute_step = [&](int buf) {
        const _Float16* aFP = aFragP + buf * A_TILE;
        const _Float16* bFP = bFragP + buf * B_TILE;
        h16v afrag[2];
#pragma unroll
        for (int mt = 0; mt < 2; mt++) {
            const _Float16* p = aFP + mt * 16 * LDS_LDA;
            h8v lo = *(const h8v*)p;
            h8v hi = *(const h8v*)(p + 8);
            afrag[mt] = __builtin_shufflevector(lo, hi, 0, 1, 2, 3, 4, 5, 6, 7,
                                                        8, 9, 10, 11, 12, 13, 14, 15);
        }
        h16v bfrag[4];
#pragma unroll
        for (int nt = 0; nt < 4; nt++)
            bfrag[nt] = *(const h16v*)(bFP + nt * 16 * LDS_LDB);
#pragma unroll
        for (int mt = 0; mt < 2; mt++)
#pragma unroll
            for (int nt = 0; nt < 4; nt++)
                acc[mt][nt] = __builtin_amdgcn_wmma_f32_16x16x32_f16(
                    false, afrag[mt], false, bfrag[nt],
                    (short)0, acc[mt][nt], false, false);
    };

    const int ksteps = (K + BK - 1) / BK;   // >= 1

    stage(0, 0);                            // prologue: tile 0 -> buf 0
    int ks = 0;
    // Steady state (branch-free): next tile's copies stay in flight across
    // the entire WMMA phase; in-order ASYNCcnt <= S proves tile ks is ready.
#pragma unroll 2
    for (; ks < ksteps - 1; ks++) {
        __syncthreads();                    // buf[(ks+1)&1]'s last readers done
        stage(ks + 1, (ks + 1) & 1);
        if constexpr (sizeof(AT) == 4)
            asm volatile("s_wait_asynccnt 0x2" ::: "memory");
        else
            asm volatile("s_wait_asynccnt 0x4" ::: "memory");
        __syncthreads();                    // tile ks visible from all waves
        compute_step(ks & 1);
    }
    // Peeled last tile: drain all async copies, then compute.
    asm volatile("s_wait_asynccnt 0x0" ::: "memory");
    __syncthreads();
    compute_step(ks & 1);

    // ---- epilogue: bias + ReLU + store ----
    const int nlocal = lane & 15;
    const int mhalf  = (lane >> 4) * 8;
#pragma unroll
    for (int mt = 0; mt < 2; mt++) {
#pragma unroll
        for (int nt = 0; nt < 4; nt++) {
            const int nglob = blockN0 + waveN * 64 + nt * 16 + nlocal;
            const float bv = (nglob < Nvalid) ? bias[nglob] : 0.0f;
#pragma unroll
            for (int r = 0; r < 8; r++) {
                const int mglob = blockM0 + waveM * 32 + mt * 16 + mhalf + r;
                float v = acc[mt][nt][r] + bv;
                v = v > 0.0f ? v : 0.0f;
                if constexpr (OUT_F32) {
                    if (mglob < M && nglob < Nvalid)
                        ((float*)Cout)[(size_t)mglob * ldc + nglob] = v;
                } else {
                    if (mglob < M)
                        ((_Float16*)Cout)[(size_t)mglob * ldc + nglob] = (_Float16)v;
                }
            }
        }
    }
}

// ---------------------------------------------------------------------------
// Launch: prep weights (f16 transposed), then 4 fused GEMM+bias+ReLU layers.
// ---------------------------------------------------------------------------
extern "C" void kernel_launch(void* const* d_in, const int* in_sizes, int n_in,
                              void* d_out, int out_size, void* d_ws, size_t ws_size,
                              hipStream_t stream)
{
    (void)in_sizes; (void)n_in; (void)out_size;

    const int M   = 100000;
    const int IN  = 3000;
    const int H   = 512;
    const int LAT = 20;
    const int INP = 3008;   // IN padded to multiple of 32

    const float* X  = (const float*)d_in[0];
    // d_in[1] = graph edges (dead input)
    const float* W0 = (const float*)d_in[2];
    const float* b0 = (const float*)d_in[3];
    const float* W1 = (const float*)d_in[4];
    const float* b1 = (const float*)d_in[5];
    const float* W2 = (const float*)d_in[6];
    const float* b2 = (const float*)d_in[7];
    const float* W3 = (const float*)d_in[8];
    const float* b3 = (const float*)d_in[9];

    // Workspace layout (256B aligned offsets)
    const size_t OFF_WT0 = 0;            // 512 x 3008 f16
    const size_t OFF_WT1 = 3080192;      // 512 x 512  f16
    const size_t OFF_WT2 = 3604480;      // 512 x 512  f16
    const size_t OFF_WT3 = 4128768;      // 128 x 512  f16 (zero rows >= 20)
    const size_t OFF_HA  = 4259840;      // 100000 x 512 f16
    const size_t OFF_HB  = 106659840;    // 100000 x 512 f16
    const size_t NEED    = 209059840;
    if (ws_size < NEED) return;

    char* ws = (char*)d_ws;
    _Float16* Wt0 = (_Float16*)(ws + OFF_WT0);
    _Float16* Wt1 = (_Float16*)(ws + OFF_WT1);
    _Float16* Wt2 = (_Float16*)(ws + OFF_WT2);
    _Float16* Wt3 = (_Float16*)(ws + OFF_WT3);
    _Float16* hA  = (_Float16*)(ws + OFF_HA);
    _Float16* hB  = (_Float16*)(ws + OFF_HB);

    {
        int t0 = H * INP;
        convert_transpose_w<<<(t0 + 255) / 256, 256, 0, stream>>>(W0, Wt0, IN, H, INP, H);
        int t1 = H * H;
        convert_transpose_w<<<(t1 + 255) / 256, 256, 0, stream>>>(W1, Wt1, H, H, H, H);
        convert_transpose_w<<<(t1 + 255) / 256, 256, 0, stream>>>(W2, Wt2, H, H, H, H);
        int t3 = 128 * H;
        convert_transpose_w<<<(t3 + 255) / 256, 256, 0, stream>>>(W3, Wt3, H, LAT, H, 128);
    }

    const dim3 blk(256);
    const dim3 gridFull((M + BM - 1) / BM, H / BN);   // 782 x 4
    const dim3 gridLast((M + BM - 1) / BM, 1);        // 782 x 1

    // L0: relu(X @ W0 + b0)   fp32 A, f16 out
    gemm_bias_relu<float, false><<<gridFull, blk, 0, stream>>>(
        X, IN, Wt0, INP, b0, hA, H, M, IN, H);
    // L1: relu(hA @ W1 + b1)
    gemm_bias_relu<_Float16, false><<<gridFull, blk, 0, stream>>>(
        hA, H, Wt1, H, b1, hB, H, M, H, H);
    // L2: relu(hB @ W2 + b2)
    gemm_bias_relu<_Float16, false><<<gridFull, blk, 0, stream>>>(
        hB, H, Wt2, H, b2, hA, H, M, H, H);
    // L3: relu(hA @ W3 + b3) -> fp32 d_out [M, 20]
    gemm_bias_relu<_Float16, true><<<gridLast, blk, 0, stream>>>(
        hA, H, Wt3, H, b3, d_out, LAT, M, H, LAT);
}
